// GINEncoder_61186104099701
// MI455X (gfx1250) — matrix operand order repaired and checked by reading, not compiled, not run on hardware
//
#include <hip/hip_runtime.h>

#define N_NODES  100000
#define N_EDGES  1600000
#define N_GRAPHS 64
#define IN_DIM   128
#define HID      128
#define EMB      64
#define LDSP     136   // 128 + 8 halves padding: 272B row stride staggers LDS banks

typedef __attribute__((ext_vector_type(16))) _Float16 v16h;
typedef __attribute__((ext_vector_type(8)))  _Float16 v8h;
typedef __attribute__((ext_vector_type(8)))  float    v8f;

union V16 { v16h v16; v8h v8[2]; };

// sigmoid with hardware reciprocal (v_rcp_f32) instead of IEEE divide:
// activation precision doesn't need the div_scale/fma Newton chain.
__device__ __forceinline__ float sigmoidf_(float x) {
    return __builtin_amdgcn_rcpf(1.0f + __expf(-x));
}

__device__ __forceinline__ void atomic_add_f32(float* p, float v) {
#if defined(__HIP_DEVICE_COMPILE__)
    unsafeAtomicAdd(p, v);   // native global_atomic_add_f32, no CAS loop
#endif
}

// One 16x16 output tile, K=128, A (f16) from LDS [16][LDSP], B (f16) from
// packed-transposed weights BT[N][128].  C initialized to broadcast bias.
// WMMA operand layouts per CDNA5 ISA 7.12.2:
//   A 16x32 f16: lane<16 holds row=lane, K = {k0..k0+7, k0+16..k0+23}
//                lane>=16 holds row=lane-16, K shifted by +8
//   B 32x16 f16: lane<16 holds col=lane, K = {k0..k0+7, k0+16..k0+23}
//                lane>=16 holds col=lane-16, K shifted by +8
__device__ __forceinline__ v8f tile_gemm128(const _Float16* __restrict__ A,
                                            const _Float16* __restrict__ BT,
                                            int n0, float bias, int hs, int l15) {
    v8f acc;
#pragma unroll
    for (int j = 0; j < 8; ++j) acc[j] = bias;
    const _Float16* ar = A + l15 * LDSP;
    const _Float16* br = BT + (n0 + l15) * 128;
#pragma unroll
    for (int k0 = 0; k0 < 128; k0 += 32) {
        V16 a, b;
        a.v8[0] = *(const v8h*)(ar + k0 + hs * 8);
        a.v8[1] = *(const v8h*)(ar + k0 + 16 + hs * 8);
        b.v8[0] = *(const v8h*)(br + k0 + hs * 8);
        b.v8[1] = *(const v8h*)(br + k0 + 16 + hs * 8);
        acc = __builtin_amdgcn_wmma_f32_16x16x32_f16(
            false, a.v16, false, b.v16, (short)0, acc, false, false);
    }
    return acc;
}

// ---- weight prep: f32 [K][N] -> f16 [N][K] -------------------------------
__global__ void pack_wT(const float* __restrict__ w, _Float16* __restrict__ wT,
                        int K, int N) {
    int i = blockIdx.x * blockDim.x + threadIdx.x;
    if (i >= K * N) return;
    int k = i / N, n = i - k * N;
    wT[n * K + k] = (_Float16)w[i];
}

// ---- edge scatter-add: one wave per edge, float4 per lane ----------------
__global__ void edge_scatter(const float* __restrict__ h,
                             const long long* __restrict__ ei,
                             float* __restrict__ agg) {
    long long g = (long long)blockIdx.x * blockDim.x + threadIdx.x;
    // e is identical across the wave: readfirstlane lets the compiler prove
    // uniformity and fetch src/dst via scalar loads (s_load, KMcnt path)
    int e = __builtin_amdgcn_readfirstlane((int)(g >> 5));
    int lane = threadIdx.x & 31;
    if (e >= N_EDGES) return;
    int src = __builtin_amdgcn_readfirstlane((int)ei[e]);
    int dst = __builtin_amdgcn_readfirstlane((int)ei[N_EDGES + e]);
    float4 v = ((const float4*)(h + (long long)src * HID))[lane];
    float* p = agg + (long long)dst * HID + lane * 4;
    atomic_add_f32(p + 0, v.x);
    atomic_add_f32(p + 1, v.y);
    atomic_add_f32(p + 2, v.z);
    atomic_add_f32(p + 3, v.w);
}

// ---- fused GIN MLP: z = h+agg; h' = relu(sigmoid(z@W1+b1)@W2+b2) ---------
__global__ void __launch_bounds__(256) gin_mlp(const float* __restrict__ hin,
                                               const float* __restrict__ agg,
                                               const _Float16* __restrict__ w1T,
                                               const float* __restrict__ b1,
                                               const _Float16* __restrict__ w2T,
                                               const float* __restrict__ b2,
                                               float* __restrict__ hout) {
    __shared__ _Float16 zsm[16 * LDSP];
    __shared__ _Float16 ysm[16 * LDSP];
    const int t = threadIdx.x;
    const long long node0 = (long long)blockIdx.x * 16;

    // stage 0: z = h + agg -> f16 LDS (16 rows x 128)
    {
        int row = t >> 4;
        int col = (t & 15) * 8;
        const float* hp = hin + (node0 + row) * HID + col;
        const float* ap = agg + (node0 + row) * HID + col;
        float4 h0 = *(const float4*)hp;
        float4 h1 = *(const float4*)(hp + 4);
        float4 a0 = *(const float4*)ap;
        float4 a1 = *(const float4*)(ap + 4);
        v8h z;
        z[0] = (_Float16)(h0.x + a0.x); z[1] = (_Float16)(h0.y + a0.y);
        z[2] = (_Float16)(h0.z + a0.z); z[3] = (_Float16)(h0.w + a0.w);
        z[4] = (_Float16)(h1.x + a1.x); z[5] = (_Float16)(h1.y + a1.y);
        z[6] = (_Float16)(h1.z + a1.z); z[7] = (_Float16)(h1.w + a1.w);
        *(v8h*)&zsm[row * LDSP + col] = z;
    }
    __syncthreads();

    const int lane = t & 31;
    const int l15  = lane & 15;
    const int hs   = lane >> 4;
    const int n0   = (t >> 5) * 16;   // wave -> 16-column tile

    // stage 1: y = sigmoid(z @ W1 + b1) -> f16 LDS
    {
        v8f acc = tile_gemm128(zsm, w1T, n0, b1[n0 + l15], hs, l15);
#pragma unroll
        for (int j = 0; j < 8; ++j)
            ysm[(hs * 8 + j) * LDSP + n0 + l15] = (_Float16)sigmoidf_(acc[j]);
    }
    __syncthreads();

    // stage 2: h' = relu(y @ W2 + b2) -> global
    {
        v8f acc = tile_gemm128(ysm, w2T, n0, b2[n0 + l15], hs, l15);
        float* op = hout + (node0 + hs * 8) * HID + n0 + l15;
#pragma unroll
        for (int j = 0; j < 8; ++j)
            op[(long long)j * HID] = fmaxf(acc[j], 0.0f);
    }
}

// ---- graph pooling scatter ------------------------------------------------
__global__ void pool_scatter(const float* __restrict__ h,
                             const long long* __restrict__ batch,
                             float* __restrict__ sums,
                             float* __restrict__ counts) {
    long long g = (long long)blockIdx.x * blockDim.x + threadIdx.x;
    int node = __builtin_amdgcn_readfirstlane((int)(g >> 5));
    int lane = threadIdx.x & 31;
    if (node >= N_NODES) return;
    int b = __builtin_amdgcn_readfirstlane((int)batch[node]);
    float4 v = ((const float4*)(h + (long long)node * HID))[lane];
    float* p = sums + b * HID + lane * 4;
    atomic_add_f32(p + 0, v.x);
    atomic_add_f32(p + 1, v.y);
    atomic_add_f32(p + 2, v.z);
    atomic_add_f32(p + 3, v.w);
    if (lane == 0) atomic_add_f32(counts + b, 1.0f);
}

// ---- pooled head: mean -> sigmoid(fc1) -> fc2 ----------------------------
__global__ void __launch_bounds__(256) pool_mlp(const float* __restrict__ sums,
                                                const float* __restrict__ counts,
                                                const _Float16* __restrict__ fc1T,
                                                const float* __restrict__ fc1b,
                                                const _Float16* __restrict__ fc2T,
                                                const float* __restrict__ fc2b,
                                                float* __restrict__ out) {
    __shared__ _Float16 psm[16 * LDSP];
    __shared__ _Float16 gsm[16 * LDSP];
    const int t = threadIdx.x;
    const int g0 = blockIdx.x * 16;

    {
        int row = t >> 4;
        int col = (t & 15) * 8;
        int g = g0 + row;
        float inv = __builtin_amdgcn_rcpf(fmaxf(counts[g], 1.0f));
        const float* sp = sums + g * HID + col;
        float4 s0 = *(const float4*)sp;
        float4 s1 = *(const float4*)(sp + 4);
        v8h z;
        z[0] = (_Float16)(s0.x * inv); z[1] = (_Float16)(s0.y * inv);
        z[2] = (_Float16)(s0.z * inv); z[3] = (_Float16)(s0.w * inv);
        z[4] = (_Float16)(s1.x * inv); z[5] = (_Float16)(s1.y * inv);
        z[6] = (_Float16)(s1.z * inv); z[7] = (_Float16)(s1.w * inv);
        *(v8h*)&psm[row * LDSP + col] = z;
    }
    __syncthreads();

    const int lane = t & 31;
    const int l15  = lane & 15;
    const int hs   = lane >> 4;
    const int wv   = t >> 5;
    const int n0   = wv * 16;

    {
        v8f acc = tile_gemm128(psm, fc1T, n0, fc1b[n0 + l15], hs, l15);
#pragma unroll
        for (int j = 0; j < 8; ++j)
            gsm[(hs * 8 + j) * LDSP + n0 + l15] = (_Float16)sigmoidf_(acc[j]);
    }
    __syncthreads();

    if (wv < 4) {  // EMB = 64 -> 4 column tiles; wave-uniform branch, EXEC stays all-ones
        v8f acc = tile_gemm128(gsm, fc2T, n0, fc2b[n0 + l15], hs, l15);
#pragma unroll
        for (int j = 0; j < 8; ++j)
            out[(g0 + hs * 8 + j) * EMB + n0 + l15] = acc[j];
    }
}

extern "C" void kernel_launch(void* const* d_in, const int* in_sizes, int n_in,
                              void* d_out, int out_size, void* d_ws, size_t ws_size,
                              hipStream_t stream) {
    const float*     x     = (const float*)d_in[0];
    const long long* ei    = (const long long*)d_in[1];
    const long long* batch = (const long long*)d_in[2];
    const float* cw1[3] = {(const float*)d_in[3],  (const float*)d_in[7],  (const float*)d_in[11]};
    const float* cb1[3] = {(const float*)d_in[4],  (const float*)d_in[8],  (const float*)d_in[12]};
    const float* cw2[3] = {(const float*)d_in[5],  (const float*)d_in[9],  (const float*)d_in[13]};
    const float* cb2[3] = {(const float*)d_in[6],  (const float*)d_in[10], (const float*)d_in[14]};
    const float* fc1_w = (const float*)d_in[15];
    const float* fc1_b = (const float*)d_in[16];
    const float* fc2_w = (const float*)d_in[17];
    const float* fc2_b = (const float*)d_in[18];

    // ---- workspace carve-up (256B aligned) ----
    char* ws = (char*)d_ws;
    size_t off = 0;
    auto alloc = [&](size_t bytes) -> void* {
        off = (off + 255) & ~(size_t)255;
        void* p = ws + off;
        off += bytes;
        return p;
    };
    const size_t featBytes = (size_t)N_NODES * HID * sizeof(float);
    float*     agg    = (float*)alloc(featBytes);
    float*     h      = (float*)alloc(featBytes);
    float*     sums   = (float*)alloc((size_t)N_GRAPHS * HID * sizeof(float));
    float*     counts = (float*)alloc((size_t)N_GRAPHS * sizeof(float));
    _Float16*  w1T[3]; _Float16* w2T[3];
    for (int l = 0; l < 3; ++l) {
        w1T[l] = (_Float16*)alloc((size_t)HID * HID * sizeof(_Float16));
        w2T[l] = (_Float16*)alloc((size_t)HID * HID * sizeof(_Float16));
    }
    _Float16* fc1T = (_Float16*)alloc((size_t)HID * HID * sizeof(_Float16));
    _Float16* fc2T = (_Float16*)alloc((size_t)HID * EMB * sizeof(_Float16));
    (void)ws_size; (void)in_sizes; (void)n_in; (void)out_size;

    // ---- pack weights to f16 transposed [N][K] ----
    const int packBlk = 256;
    for (int l = 0; l < 3; ++l) {
        pack_wT<<<(HID * HID + packBlk - 1) / packBlk, packBlk, 0, stream>>>(cw1[l], w1T[l], HID, HID);
        pack_wT<<<(HID * HID + packBlk - 1) / packBlk, packBlk, 0, stream>>>(cw2[l], w2T[l], HID, HID);
    }
    pack_wT<<<(HID * HID + packBlk - 1) / packBlk, packBlk, 0, stream>>>(fc1_w, fc1T, HID, HID);
    pack_wT<<<(HID * EMB + packBlk - 1) / packBlk, packBlk, 0, stream>>>(fc2_w, fc2T, HID, EMB);

    const int scatterGrid = (N_EDGES * 32) / 256;         // 200000 blocks
    const int mlpGrid     = N_NODES / 16;                 // 6250 blocks
    const int poolGrid    = (N_NODES * 32 + 255) / 256;   // 12500 blocks

    // ---- 3 GIN layers ----
    for (int l = 0; l < 3; ++l) {
        const float* hin = (l == 0) ? x : h;
        hipMemsetAsync(agg, 0, featBytes, stream);
        edge_scatter<<<scatterGrid, 256, 0, stream>>>(hin, ei, agg);
        gin_mlp<<<mlpGrid, 256, 0, stream>>>(hin, agg, w1T[l], cb1[l], w2T[l], cb2[l], h);
    }

    // ---- global mean pool + head ----
    hipMemsetAsync(sums, 0, (size_t)N_GRAPHS * HID * sizeof(float), stream);
    hipMemsetAsync(counts, 0, (size_t)N_GRAPHS * sizeof(float), stream);
    pool_scatter<<<poolGrid, 256, 0, stream>>>(h, batch, sums, counts);
    pool_mlp<<<N_GRAPHS / 16, 256, 0, stream>>>(sums, counts, fc1T, fc1_b, fc2T, fc2_b, (float*)d_out);
}
